// VanillaRNN_21088289423738
// MI455X (gfx1250) — matrix operand order repaired and checked by reading, not compiled, not run on hardware
//
#include <hip/hip_runtime.h>
#include <hip/hip_bf16.h>
#include <math.h>

// Problem constants (from reference): B=1024, S=128, H=1024, C=10
#define RNN_B 1024
#define RNN_S 128
#define RNN_H 1024
#define RNN_C 10

typedef __attribute__((ext_vector_type(16))) _Float16 v16h;
typedef __attribute__((ext_vector_type(8)))  _Float16 v8h;
typedef __attribute__((ext_vector_type(8)))  float    v8f;
typedef __attribute__((ext_vector_type(4)))  int      v4i;

// --- CDNA5 async global->LDS path (guarded; falls back to register staging) --
#if defined(__gfx1250__) && __has_builtin(__builtin_amdgcn_global_load_async_to_lds_b128)
#define USE_ASYNC_LDS 1
typedef __attribute__((address_space(1))) v4i* gptr_v4i;   // global
typedef __attribute__((address_space(3))) v4i* lptr_v4i;   // LDS
#endif

__device__ __forceinline__ void g2lds16(_Float16* lds, const _Float16* g) {
#ifdef USE_ASYNC_LDS
    __builtin_amdgcn_global_load_async_to_lds_b128(
        (gptr_v4i)(_Float16*)g,       // src (global), const cast away
        (lptr_v4i)lds,                // dst (LDS)
        /*offset=*/0, /*cpol=*/0);
#else
    *(v8h*)lds = *(const v8h*)g;
#endif
}

__device__ __forceinline__ void wait_async_done() {
#ifdef USE_ASYNC_LDS
#if __has_builtin(__builtin_amdgcn_s_wait_asynccnt)
    __builtin_amdgcn_s_wait_asynccnt(0);
#else
    asm volatile("s_wait_asynccnt 0" ::: "memory");
#endif
#endif
}

// Branchless tanh: 1 - 2/(exp(2v)+1).  Exact at +/-inf, ~1 ulp f32 — far
// beyond f16 storage precision.  v_exp_f32 + v_rcp_f32, no exec-mask branches.
__device__ __forceinline__ float fast_tanh(float v) {
    float e = __builtin_amdgcn_exp2f(v * 2.88539008177793f);   // exp(2v)
    float r = __builtin_amdgcn_rcpf(e + 1.0f);
    return __builtin_fmaf(-2.0f, r, 1.0f);
}

// ---------------------------------------------------------------------------
// Kernel 1: convert whh (f32, HxH row-major) -> f16, same layout.
// Row-major whh[j][k] is exactly the WMMA B-operand tile layout [n][k].
// ---------------------------------------------------------------------------
__global__ __launch_bounds__(256) void k_convert_whh(const float* __restrict__ whh,
                                                     _Float16* __restrict__ w16) {
    int i = blockIdx.x * 256 + threadIdx.x;           // H*H / 256 blocks, exact
    w16[i] = (_Float16)whh[i];
}

// ---------------------------------------------------------------------------
// Kernel 2: step t=0 with h=0:  h1 = tanh(x[:,0] * whx + bh), stored f16.
// ---------------------------------------------------------------------------
__global__ __launch_bounds__(256) void k_step0(const float* __restrict__ x,
                                               const float* __restrict__ whx,
                                               const float* __restrict__ bh,
                                               _Float16* __restrict__ hout) {
    int i = blockIdx.x * 256 + threadIdx.x;           // B*H / 256 blocks, exact
    int b = i >> 10;                                  // /H
    int j = i & (RNN_H - 1);
    hout[i] = (_Float16)fast_tanh(x[b * RNN_S] * whx[j] + bh[j]);
}

// ---------------------------------------------------------------------------
// Kernel 3: one recurrent step.
//   hout[b][n] = tanh( sum_k hin[b][k]*w16[n][k]  + x[b][t]*whx[n] + bh[n] )
// Block tile: 128(M) x 64(N), K-chunks of 32, double-buffered LDS with
// async global->LDS prefetch of tile k+1 overlapping WMMA on tile k.
// 256 threads = 8 waves in a 4x2 grid, each wave 32x32 -> 2x2 WMMA frags.
// ---------------------------------------------------------------------------
__global__ __launch_bounds__(256) void k_rnn_step(const _Float16* __restrict__ hin,
                                                  _Float16* __restrict__ hout,
                                                  const _Float16* __restrict__ w16,
                                                  const float* __restrict__ x,
                                                  const float* __restrict__ whx,
                                                  const float* __restrict__ bh,
                                                  int t) {
    constexpr int H = RNN_H, S = RNN_S;
    constexpr int LDA = 40;  // 32 + 8 halves pad; rows are 80B (16B-aligned)
    __shared__ __attribute__((aligned(16))) _Float16 As[2][128 * LDA]; // h tile [m][k]
    __shared__ __attribute__((aligned(16))) _Float16 Bs[2][64 * LDA];  // w tile [n][k]

    const int tid  = threadIdx.x;
    const int lane = tid & 31;
    const int wid  = tid >> 5;
    const int lh   = lane & 15;          // lane within half-wave
    const int sel  = lane >> 4;          // which half-wave
    const int m_off = (wid >> 1) * 32;   // wave M offset inside block tile
    const int n_off = (wid & 1) * 32;    // wave N offset inside block tile
    const int blockM = blockIdx.x * 128;
    const int blockN = blockIdx.y * 64;

    // Per-thread staging coordinates (16B chunks of 8 halves).
    const int rA0 = tid >> 2,          cA0 = (tid & 3) * 8;         // A chunk 0
    const int rA1 = (tid + 256) >> 2,  cA1 = cA0;                   // A chunk 1
    const int rB  = tid >> 2,          cB  = cA0;                   // B chunk

    v8f acc[2][2] = {};

    auto stage = [&](int buf, int k0) {
        g2lds16(&As[buf][rA0 * LDA + cA0], &hin[(blockM + rA0) * H + k0 + cA0]);
        g2lds16(&As[buf][rA1 * LDA + cA1], &hin[(blockM + rA1) * H + k0 + cA1]);
        g2lds16(&Bs[buf][rB  * LDA + cB ], &w16[(blockN + rB ) * H + k0 + cB ]);
    };

    // Prologue: stage tile 0.
    stage(0, 0);
    wait_async_done();
    __syncthreads();

    int cur = 0;
    for (int k0 = 0; k0 < H; k0 += 32) {
        const bool more = (k0 + 32) < H;
        if (more) stage(cur ^ 1, k0 + 32);     // prefetch next tile into other buffer

        // ---- fragments per documented 16-bit WMMA lane layouts ----
        // A 16x32: lane lh = row M; sel picks K {0..7,16..23} vs {8..15,24..31}
        v16h a[2], b[2];
        #pragma unroll
        for (int mi = 0; mi < 2; ++mi) {
            int row = m_off + mi * 16 + lh;
            v8h lo = *(const v8h*)&As[cur][row * LDA + sel * 8];
            v8h hi = *(const v8h*)&As[cur][row * LDA + sel * 8 + 16];
            a[mi] = __builtin_shufflevector(lo, hi, 0,1,2,3,4,5,6,7,8,9,10,11,12,13,14,15);
        }
        // B 32x16: lane lh = col N; sel picks K {0..7,16..23} vs {8..15,24..31}
        #pragma unroll
        for (int ni = 0; ni < 2; ++ni) {
            int col = n_off + ni * 16 + lh;
            v8h lo = *(const v8h*)&Bs[cur][col * LDA + sel * 8];
            v8h hi = *(const v8h*)&Bs[cur][col * LDA + sel * 8 + 16];
            b[ni] = __builtin_shufflevector(lo, hi, 0,1,2,3,4,5,6,7,8,9,10,11,12,13,14,15);
        }

        #pragma unroll
        for (int mi = 0; mi < 2; ++mi)
            #pragma unroll
            for (int ni = 0; ni < 2; ++ni)
                acc[mi][ni] = __builtin_amdgcn_wmma_f32_16x16x32_f16(
                    false, a[mi], false, b[ni], (short)0, acc[mi][ni], false, false);

        if (more) {
            wait_async_done();   // our prefetch chunks are in LDS
            __syncthreads();     // everyone done reading cur + done writing cur^1
            cur ^= 1;
        }
    }

    // ---- fused epilogue: tanh(acc + x_t*whx + bh), store f16 ----
    // D layout: lane lh = N col; VGPR r -> M = sel*8 + r
    #pragma unroll
    for (int ni = 0; ni < 2; ++ni) {
        int n = blockN + n_off + ni * 16 + lh;
        float whx_n = whx[n];
        float bh_n  = bh[n];
        #pragma unroll
        for (int mi = 0; mi < 2; ++mi) {
            #pragma unroll
            for (int r = 0; r < 8; ++r) {
                int m = blockM + m_off + mi * 16 + sel * 8 + r;
                float v = acc[mi][ni][r] + x[m * S + t] * whx_n + bh_n;
                hout[m * H + n] = (_Float16)fast_tanh(v);
            }
        }
    }
}

// ---------------------------------------------------------------------------
// Kernel 4: projection out[b][c] = sum_j h[b][j]*wph[c][j] + bp[c].
// One wave per batch row; 10 accumulators per lane; shfl_xor reduction.
// ---------------------------------------------------------------------------
__global__ __launch_bounds__(256) void k_proj(const _Float16* __restrict__ h,
                                              const float* __restrict__ wph,
                                              const float* __restrict__ bp,
                                              float* __restrict__ out) {
    const int lane = threadIdx.x & 31;
    const int wid  = threadIdx.x >> 5;
    const int b    = blockIdx.x * 8 + wid;            // 128 blocks x 8 waves = 1024

    float acc[RNN_C];
    #pragma unroll
    for (int c = 0; c < RNN_C; ++c) acc[c] = 0.0f;

    for (int j = lane; j < RNN_H; j += 32) {
        float hv = (float)h[b * RNN_H + j];
        #pragma unroll
        for (int c = 0; c < RNN_C; ++c)
            acc[c] += hv * wph[c * RNN_H + j];
    }
    #pragma unroll
    for (int c = 0; c < RNN_C; ++c) {
        float v = acc[c];
        #pragma unroll
        for (int m = 16; m >= 1; m >>= 1)
            v += __shfl_xor(v, m, 32);
        if (lane == 0) out[b * RNN_C + c] = v + bp[c];
    }
}

// ---------------------------------------------------------------------------
// Host side
// ---------------------------------------------------------------------------
extern "C" void kernel_launch(void* const* d_in, const int* in_sizes, int n_in,
                              void* d_out, int out_size, void* d_ws, size_t ws_size,
                              hipStream_t stream) {
    const float* x   = (const float*)d_in[0];   // (B,S)
    const float* whx = (const float*)d_in[1];   // (H,1)
    const float* whh = (const float*)d_in[2];   // (H,H)
    const float* bh  = (const float*)d_in[3];   // (H,)
    const float* wph = (const float*)d_in[4];   // (C,H)
    const float* bp  = (const float*)d_in[5];   // (C,)
    float* out = (float*)d_out;

    char* ws = (char*)d_ws;
    const size_t szW = (size_t)RNN_H * RNN_H * sizeof(_Float16); // 2 MB
    const size_t szH = (size_t)RNN_B * RNN_H * sizeof(_Float16); // 2 MB
    _Float16* w16 = (_Float16*)(ws);
    _Float16* hA  = (_Float16*)(ws + szW);
    _Float16* hB  = (_Float16*)(ws + szW + szH);

    // One-time (per launch) weight conversion; deterministic.
    k_convert_whh<<<(RNN_H * RNN_H) / 256, 256, 0, stream>>>(whh, w16);

    // t = 0 (h starts at zero, GEMM degenerates)
    k_step0<<<(RNN_B * RNN_H) / 256, 256, 0, stream>>>(x, whx, bh, hA);

    _Float16* cur = hA;
    _Float16* nxt = hB;
    dim3 grid(RNN_B / 128, RNN_H / 64);
    for (int t = 1; t < RNN_S; ++t) {
        k_rnn_step<<<grid, 256, 0, stream>>>(cur, nxt, w16, x, whx, bh, t);
        _Float16* tmp = cur; cur = nxt; nxt = tmp;
    }

    k_proj<<<RNN_B / 8, 256, 0, stream>>>(cur, wph, bp, out);
}
